// MultiHeadAttention_77644418777867
// MI455X (gfx1250) — compile-verified, matrix operand order checked by reading, not run on hardware
//
#include <hip/hip_runtime.h>
#include <stdint.h>

#define D_MODEL   1024
#define NUM_HEADS 16
#define D_K       64
#define SEQ       2048
#define BATCH     2

typedef __attribute__((ext_vector_type(16))) __bf16 v16bf;
typedef __attribute__((ext_vector_type(8)))  float  v8f;

struct alignas(16) U128 { unsigned int x, y, z, w; };
struct alignas(8)  U64  { unsigned int x, y; };
struct alignas(16) F128 { float x, y, z, w; };

union FragU { U128 u[2]; v16bf v; };
union V8U   { U128 u; unsigned short s[8]; };

// ---- bf16 conversion: prefer v_cvt_pk_bf16_f32 if the toolchain has it ----
#if __has_builtin(__builtin_amdgcn_cvt_pk_bf16_f32)
typedef __attribute__((ext_vector_type(2))) __bf16 v2bf;
__device__ inline unsigned int f2bf2(float lo, float hi) {
  union { v2bf v; unsigned int u; } c;
  c.v = __builtin_amdgcn_cvt_pk_bf16_f32(lo, hi);
  return c.u;
}
__device__ inline unsigned short f2bf(float f) {
  return (unsigned short)(f2bf2(f, 0.0f) & 0xffffu);
}
#else
__device__ inline unsigned short f2bf(float f) {
  unsigned int u = __float_as_uint(f);
  u += 0x7FFFu + ((u >> 16) & 1u);          // round-to-nearest-even
  return (unsigned short)(u >> 16);
}
__device__ inline unsigned int f2bf2(float lo, float hi) {
  return (unsigned int)f2bf(lo) | ((unsigned int)f2bf(hi) << 16);
}
#endif

__device__ inline v8f v8f_zero() {
  v8f z;
#pragma unroll
  for (int i = 0; i < 8; ++i) z[i] = 0.0f;
  return z;
}

__device__ inline v8f wmma_bf16(v16bf a, v16bf b, v8f c) {
  // D(f32 16x16) = A(bf16 16x32) * B(bf16 32x16) + C
  return __builtin_amdgcn_wmma_f32_16x16x32_bf16(false, a, false, b, (short)0, c,
                                                 false, false);
}

// Load one 16x32 bf16 operand fragment (A- or B-style) from LDS/global memory.
// CDNA5 16-bit A-matrix layout: lane L holds row (row0 + L%16); per-lane halves
// are [kb..kb+7] and [kb+16..kb+23] with kb = kofs + 8*(L>=16).
__device__ inline v16bf load_frag(const unsigned short* p, int stride, int row0, int kofs) {
  const int lane = threadIdx.x & 31;
  const unsigned short* r = p + (row0 + (lane & 15)) * stride + kofs + ((lane >> 4) << 3);
  FragU f;
  f.u[0] = *reinterpret_cast<const U128*>(r);
  f.u[1] = *reinterpret_cast<const U128*>(r + 16);
  return f.v;
}

__device__ inline float rmax16(float v) {
#pragma unroll
  for (int m = 1; m < 16; m <<= 1) v = fmaxf(v, __shfl_xor(v, m, 32));
  return v;
}
__device__ inline float rsum16(float v) {
#pragma unroll
  for (int m = 1; m < 16; m <<= 1) v += __shfl_xor(v, m, 32);
  return v;
}

// ---------------------------------------------------------------------------
// Tensor Data Mover: 2D bf16 tile Global -> LDS with row padding.
// This toolchain declares the 6-arg builtin:
//   (uint32x4 g0, int32x8 g1, int32x4, int32x4, int32x8, i32 cpol)
// Groups 2/3 (and the trailing group) are zero / unused for 2D tensors.
// ---------------------------------------------------------------------------
#if __has_builtin(__builtin_amdgcn_tensor_load_to_lds) && \
    __has_builtin(__builtin_amdgcn_s_wait_tensorcnt)
#define USE_TDM 1
typedef unsigned int tdm_u4 __attribute__((ext_vector_type(4)));
typedef int          tdm_i8 __attribute__((ext_vector_type(8)));
typedef int          tdm_i4 __attribute__((ext_vector_type(4)));

// tile_w/tensor_w/row_stride in bf16 elements; pad codes per D# encoding:
// pad_interval k -> pad after 2^(k+1) DWORDs; pad_amount k -> (k+1) DWORDs.
__device__ inline void tdm_load_2d_bf16(unsigned lds_off, const void* gaddr,
                                        unsigned tile_w, unsigned tile_h,
                                        unsigned tensor_w, unsigned tensor_h,
                                        unsigned row_stride,
                                        unsigned pad_interval_code,
                                        unsigned pad_amount_code) {
  const unsigned long long ga = (unsigned long long)(uintptr_t)gaddr;
  tdm_u4 g0;
  g0[0] = 1u;                                               // count=1, user D#
  g0[1] = lds_off;                                          // lds_addr
  g0[2] = (unsigned)(ga & 0xffffffffu);                     // global_addr[31:0]
  g0[3] = (unsigned)((ga >> 32) & 0x01ffffffu) | (2u << 30);// addr[56:32]|type=2
  tdm_i8 g1;
  g1[0] = (int)((1u << 16) |                                // data_size = 2B
                (1u << 20) |                                // pad_enable
                (pad_interval_code << 22) | (pad_amount_code << 25));
  g1[1] = (int)((tensor_w & 0xffffu) << 16);                // tensor_dim0 lo
  g1[2] = (int)((tensor_w >> 16) | ((tensor_h & 0xffffu) << 16));
  g1[3] = (int)((tensor_h >> 16) | (tile_w << 16));         // tile_dim0
  g1[4] = (int)tile_h;                                      // tile_dim1 (dim2=0)
  g1[5] = (int)row_stride;                                  // tensor_dim0_stride
  g1[6] = 0;
  g1[7] = 0;
  tdm_i4 gz4;
  gz4[0] = 0; gz4[1] = 0; gz4[2] = 0; gz4[3] = 0;
  tdm_i8 gz8;
#pragma unroll
  for (int i = 0; i < 8; ++i) gz8[i] = 0;
  __builtin_amdgcn_tensor_load_to_lds(g0, g1, gz4, gz4, gz8, 0);
}
#define TDM_WAIT() __builtin_amdgcn_s_wait_tensorcnt(0)
#else
#define USE_TDM 0
#define TDM_WAIT()
#endif

// ---------------------------------------------------------------------------
// Kernel 1: fused Q/K/V projection.  Y = X @ W.T + b, written bf16 into
// per-head layout [B, H, S, dk].  blockIdx.z picks which of Q/K/V.
// Block = 128 threads (4 waves), tile 64(M) x 64(N), K-step 32.
// ---------------------------------------------------------------------------
__global__ __launch_bounds__(128) void mha_qkv_proj(
    const float* __restrict__ q, const float* __restrict__ k, const float* __restrict__ v,
    const float* __restrict__ wq, const float* __restrict__ wk, const float* __restrict__ wv,
    const float* __restrict__ bq, const float* __restrict__ bk, const float* __restrict__ bv,
    unsigned short* __restrict__ qh, unsigned short* __restrict__ kh, unsigned short* __restrict__ vh)
{
  __shared__ __align__(16) unsigned short As[64 * 40];   // 64 rows x 32 halves (+8 pad)
  __shared__ __align__(16) unsigned short Ws[64 * 40];

  const int which = blockIdx.z;
  const float* A    = (which == 0) ? q  : (which == 1) ? k  : v;
  const float* W    = (which == 0) ? wq : (which == 1) ? wk : wv;
  const float* bias = (which == 0) ? bq : (which == 1) ? bk : bv;
  unsigned short* O = (which == 0) ? qh : (which == 1) ? kh : vh;

  const int h   = blockIdx.x;      // head index == 64-wide N block
  const int bm  = blockIdx.y;      // 64-row M block
  const int tid = threadIdx.x;
  const int wave = tid >> 5;
  const int lane = tid & 31;

  v8f acc[4];
#pragma unroll
  for (int nt = 0; nt < 4; ++nt) acc[nt] = v8f_zero();

  for (int k0 = 0; k0 < D_MODEL; k0 += 32) {
#pragma unroll
    for (int i = 0; i < 4; ++i) {
      const int idx = tid + i * 128;            // 0..511 : 64 rows x 8 float4-chunks
      const int row = idx >> 3;
      const int c4  = (idx & 7) << 2;
      F128 af = *reinterpret_cast<const F128*>(A + (size_t)(bm * 64 + row) * D_MODEL + k0 + c4);
      U64 ap; ap.x = f2bf2(af.x, af.y); ap.y = f2bf2(af.z, af.w);
      *reinterpret_cast<U64*>(As + row * 40 + c4) = ap;
      F128 wf = *reinterpret_cast<const F128*>(W + (size_t)(h * 64 + row) * D_MODEL + k0 + c4);
      U64 wp; wp.x = f2bf2(wf.x, wf.y); wp.y = f2bf2(wf.z, wf.w);
      *reinterpret_cast<U64*>(Ws + row * 40 + c4) = wp;
    }
    if (k0 + 32 < D_MODEL) {                    // gfx1250 global_prefetch_b8
      __builtin_prefetch(A + (size_t)(bm * 64 + (tid >> 1)) * D_MODEL + k0 + 32 + (tid & 1) * 16, 0, 1);
      __builtin_prefetch(W + (size_t)(h  * 64 + (tid >> 1)) * D_MODEL + k0 + 32 + (tid & 1) * 16, 0, 1);
    }
    __syncthreads();
    v16bf afrag = load_frag(As, 40, wave * 16, 0);
#pragma unroll
    for (int nt = 0; nt < 4; ++nt) {
      v16bf bfrag = load_frag(Ws, 40, nt * 16, 0);
      acc[nt] = wmma_bf16(afrag, bfrag, acc[nt]);
    }
    __syncthreads();
  }

  // Epilogue: + bias, bf16, head layout [(b*H+h)*S + s]*64 + d
  const int half = lane >> 4;
  const int col  = lane & 15;
#pragma unroll
  for (int nt = 0; nt < 4; ++nt) {
    const int d = nt * 16 + col;
    const float bb = bias[h * 64 + d];
#pragma unroll
    for (int r = 0; r < 8; ++r) {
      const int m = bm * 64 + wave * 16 + r + half * 8;
      const int b = m >> 11;                 // m / SEQ
      const int s = m & (SEQ - 1);
      O[((size_t)(b * NUM_HEADS + h) * SEQ + s) * D_K + d] = f2bf(acc[nt][r] + bb);
    }
  }
}

// ---------------------------------------------------------------------------
// Kernel 2: flash attention over one (b, h, 64-row Q block).
// 4 waves, each owns 16 Q rows; loop over 64-key blocks with online softmax.
// K tile streamed by the Tensor Data Mover (wave 0 issues, TENSORcnt waits).
// ---------------------------------------------------------------------------
__global__ __launch_bounds__(128) void mha_attention(
    const unsigned short* __restrict__ qh,
    const unsigned short* __restrict__ kh,
    const unsigned short* __restrict__ vh,
    unsigned short* __restrict__ attn)
{
  __shared__ __align__(16) unsigned short Ks[64 * 72];       // [key][d]
  __shared__ __align__(16) unsigned short Vs[64 * 72];       // [d][key]  (transposed)
  __shared__ __align__(16) unsigned short Ps[4 * 16 * 72];   // per-wave P staging

  const int qb  = blockIdx.x;          // 0..SEQ/64-1
  const int h   = blockIdx.y;
  const int b   = blockIdx.z;
  const int tid = threadIdx.x;
  const int wave = tid >> 5;
  const int lane = tid & 31;

  const size_t headoff = (size_t)(b * NUM_HEADS + h) * SEQ * D_K;
  const unsigned short* Qp = qh + headoff;
  const unsigned short* Kp = kh + headoff;
  const unsigned short* Vp = vh + headoff;

  // Q fragments for this wave's 16 rows: two 16x32 chunks over d.
  const unsigned short* qbase = Qp + (size_t)(qb * 64 + wave * 16) * D_K;
  const v16bf qfrag0 = load_frag(qbase, D_K, 0, 0);
  const v16bf qfrag1 = load_frag(qbase, D_K, 0, 32);

  float mrow[8], lrow[8];
  v8f Oacc[4];
#pragma unroll
  for (int r = 0; r < 8; ++r) { mrow[r] = -1e30f; lrow[r] = 0.0f; }
#pragma unroll
  for (int dt = 0; dt < 4; ++dt) Oacc[dt] = v8f_zero();

  const float sc = 0.125f * 1.44269504088896340736f;   // 1/sqrt(dk) * log2(e)

#if USE_TDM
  const unsigned ks_off = (unsigned)(uintptr_t)(void*)Ks;   // LDS byte offset
#endif

  for (int kt = 0; kt < SEQ / 64; ++kt) {
#if USE_TDM
    // K tile [64 keys x 64 d] -> LDS rows padded 64->72 halves, one TDM op.
    if (wave == 0) {
      tdm_load_2d_bf16(ks_off, Kp + (size_t)kt * 64 * D_K,
                       /*tile_w=*/64, /*tile_h=*/64,
                       /*tensor_w=*/D_K, /*tensor_h=*/SEQ,
                       /*row_stride=*/D_K,
                       /*pad_interval(32 DW)=*/4, /*pad_amount(4 DW)=*/3);
    }
#endif
    // V tile loaded + transposed into [d][key] by all waves.
#pragma unroll
    for (int i = 0; i < 4; ++i) {
      const int idx = tid + i * 128;          // 64 rows x 8 16B-chunks
      const int row = idx >> 3;
      const int c8  = (idx & 7) << 3;
      const size_t goff = (size_t)(kt * 64 + row) * D_K + c8;
#if !USE_TDM
      *reinterpret_cast<U128*>(Ks + row * 72 + c8) =
          *reinterpret_cast<const U128*>(Kp + goff);
#endif
      V8U vv; vv.u = *reinterpret_cast<const U128*>(Vp + goff);
#pragma unroll
      for (int j = 0; j < 8; ++j) Vs[(c8 + j) * 72 + row] = vv.s[j];
    }
    if (kt + 1 < SEQ / 64) {
      const size_t noff = (size_t)(kt + 1) * 64 * D_K + (size_t)tid * 32;
      __builtin_prefetch(Vp + noff, 0, 1);
#if !USE_TDM
      __builtin_prefetch(Kp + noff, 0, 1);
#endif
    }
#if USE_TDM
    if (wave == 0) TDM_WAIT();
#endif
    __syncthreads();

    // Scores: S = Q(16x64) . K(64-keys)^T  -> 4 tiles of 16x16, fp32.
    v8f sacc[4];
#pragma unroll
    for (int nt = 0; nt < 4; ++nt) {
      sacc[nt] = v8f_zero();
      v16bf kf0 = load_frag(Ks, 72, nt * 16, 0);
      sacc[nt] = wmma_bf16(qfrag0, kf0, sacc[nt]);
      v16bf kf1 = load_frag(Ks, 72, nt * 16, 32);
      sacc[nt] = wmma_bf16(qfrag1, kf1, sacc[nt]);
    }

    // Online softmax (base-2).  C layout: VGPR r <-> row r+8*(lane>=16),
    // lane%16 <-> column, so 16-lane shuffles reduce across keys.
    float pr[4][8];
#pragma unroll
    for (int r = 0; r < 8; ++r) {
      float mx = -1e30f;
#pragma unroll
      for (int nt = 0; nt < 4; ++nt) {
        const float s2 = sacc[nt][r] * sc;
        pr[nt][r] = s2;
        mx = fmaxf(mx, s2);
      }
      mx = rmax16(mx);
      const float mnew  = fmaxf(mrow[r], mx);
      const float alpha = exp2f(mrow[r] - mnew);
      mrow[r] = mnew;
      float rs = 0.0f;
#pragma unroll
      for (int nt = 0; nt < 4; ++nt) {
        const float p = exp2f(pr[nt][r] - mnew);
        pr[nt][r] = p;
        rs += p;
      }
      rs = rsum16(rs);
      lrow[r] = lrow[r] * alpha + rs;
#pragma unroll
      for (int dt = 0; dt < 4; ++dt) Oacc[dt][r] *= alpha;
    }

    // Re-swizzle P (C layout -> A fragment) through per-wave LDS.
    unsigned short* Pw = Ps + wave * 16 * 72;
#pragma unroll
    for (int nt = 0; nt < 4; ++nt)
#pragma unroll
      for (int r = 0; r < 8; ++r)
        Pw[(r + 8 * (lane >> 4)) * 72 + nt * 16 + (lane & 15)] = f2bf(pr[nt][r]);

    const v16bf pf0 = load_frag(Pw, 72, 0, 0);
    const v16bf pf1 = load_frag(Pw, 72, 0, 32);
#pragma unroll
    for (int dt = 0; dt < 4; ++dt) {
      v16bf vf0 = load_frag(Vs, 72, dt * 16, 0);
      Oacc[dt] = wmma_bf16(pf0, vf0, Oacc[dt]);
      v16bf vf1 = load_frag(Vs, 72, dt * 16, 32);
      Oacc[dt] = wmma_bf16(pf1, vf1, Oacc[dt]);
    }
    __syncthreads();
  }

  // Normalize and store bf16 into attn[b, s, h*64 + d] (= transpose+reshape).
#pragma unroll
  for (int r = 0; r < 8; ++r) {
    const float inv = 1.0f / lrow[r];
    const int s = qb * 64 + wave * 16 + r + 8 * (lane >> 4);
#pragma unroll
    for (int dt = 0; dt < 4; ++dt) {
      const int n = h * 64 + dt * 16 + (lane & 15);
      attn[(size_t)(b * SEQ + s) * D_MODEL + n] = f2bf(Oacc[dt][r] * inv);
    }
  }
}

// ---------------------------------------------------------------------------
// Kernel 3: output projection.  out = attn(bf16) @ wo.T + bo  (fp32 out).
// A tile (already bf16) streamed by TDM; W tile converted manually.
// ---------------------------------------------------------------------------
__global__ __launch_bounds__(128) void mha_out_proj(
    const unsigned short* __restrict__ attn, const float* __restrict__ wo,
    const float* __restrict__ bo, float* __restrict__ out)
{
  __shared__ __align__(16) unsigned short As[64 * 40];
  __shared__ __align__(16) unsigned short Ws[64 * 40];

  const int bn  = blockIdx.x;      // 64-wide N block
  const int bm  = blockIdx.y;      // 64-row M block
  const int tid = threadIdx.x;
  const int wave = tid >> 5;
  const int lane = tid & 31;

  v8f acc[4];
#pragma unroll
  for (int nt = 0; nt < 4; ++nt) acc[nt] = v8f_zero();

#if USE_TDM
  const unsigned as_off = (unsigned)(uintptr_t)(void*)As;
#endif

  for (int k0 = 0; k0 < D_MODEL; k0 += 32) {
#if USE_TDM
    // A tile [64 rows x 32 halves], rows padded 32->40 halves, one TDM op.
    if (wave == 0) {
      tdm_load_2d_bf16(as_off, attn + (size_t)(bm * 64) * D_MODEL + k0,
                       /*tile_w=*/32, /*tile_h=*/64,
                       /*tensor_w=*/D_MODEL, /*tensor_h=*/BATCH * SEQ,
                       /*row_stride=*/D_MODEL,
                       /*pad_interval(16 DW)=*/3, /*pad_amount(4 DW)=*/3);
    }
#endif
#pragma unroll
    for (int i = 0; i < 4; ++i) {
      const int idx = tid + i * 128;
      const int row = idx >> 3;
      const int c4  = (idx & 7) << 2;
#if !USE_TDM
      *reinterpret_cast<U64*>(As + row * 40 + c4) =
          *reinterpret_cast<const U64*>(attn + (size_t)(bm * 64 + row) * D_MODEL + k0 + c4);
#endif
      F128 wf = *reinterpret_cast<const F128*>(wo + (size_t)(bn * 64 + row) * D_MODEL + k0 + c4);
      U64 wp; wp.x = f2bf2(wf.x, wf.y); wp.y = f2bf2(wf.z, wf.w);
      *reinterpret_cast<U64*>(Ws + row * 40 + c4) = wp;
    }
    if (k0 + 32 < D_MODEL) {
      __builtin_prefetch(wo + (size_t)(bn * 64 + (tid >> 1)) * D_MODEL + k0 + 32 + (tid & 1) * 16, 0, 1);
#if !USE_TDM
      __builtin_prefetch(attn + (size_t)(bm * 64 + (tid >> 1)) * D_MODEL + k0 + 32 + (tid & 1) * 16, 0, 1);
#endif
    }
#if USE_TDM
    if (wave == 0) TDM_WAIT();
#endif
    __syncthreads();
    v16bf afrag = load_frag(As, 40, wave * 16, 0);
#pragma unroll
    for (int nt = 0; nt < 4; ++nt) {
      v16bf bfrag = load_frag(Ws, 40, nt * 16, 0);
      acc[nt] = wmma_bf16(afrag, bfrag, acc[nt]);
    }
    __syncthreads();
  }

  const int half = lane >> 4;
  const int col  = lane & 15;
#pragma unroll
  for (int nt = 0; nt < 4; ++nt) {
    const int n = bn * 64 + nt * 16 + col;
    const float bb = bo[n];
#pragma unroll
    for (int r = 0; r < 8; ++r) {
      const int m = bm * 64 + wave * 16 + r + half * 8;
      out[(size_t)m * D_MODEL + n] = acc[nt][r] + bb;
    }
  }
}

// ---------------------------------------------------------------------------
extern "C" void kernel_launch(void* const* d_in, const int* in_sizes, int n_in,
                              void* d_out, int out_size, void* d_ws, size_t ws_size,
                              hipStream_t stream) {
  (void)in_sizes; (void)n_in; (void)out_size; (void)ws_size;
  const float* q  = (const float*)d_in[0];
  const float* k  = (const float*)d_in[1];
  const float* v  = (const float*)d_in[2];
  const float* wq = (const float*)d_in[3];
  const float* bq = (const float*)d_in[4];
  const float* wk = (const float*)d_in[5];
  const float* bk = (const float*)d_in[6];
  const float* wv = (const float*)d_in[7];
  const float* bv = (const float*)d_in[8];
  const float* wo = (const float*)d_in[9];
  const float* bo = (const float*)d_in[10];
  float* out = (float*)d_out;

  unsigned short* ws = (unsigned short*)d_ws;
  const size_t HELE = (size_t)BATCH * NUM_HEADS * SEQ * D_K;   // 4M bf16 elems
  unsigned short* qh    = ws;
  unsigned short* kh    = ws + HELE;
  unsigned short* vh    = ws + 2 * HELE;
  unsigned short* attnb = ws + 3 * HELE;                        // total 32 MB

  const int MBLK = (BATCH * SEQ) / 64;   // 64

  mha_qkv_proj<<<dim3(NUM_HEADS, MBLK, 3), 128, 0, stream>>>(
      q, k, v, wq, wk, wv, bq, bk, bv, qh, kh, vh);
  mha_attention<<<dim3(SEQ / 64, NUM_HEADS, BATCH), 128, 0, stream>>>(
      qh, kh, vh, attnb);
  mha_out_proj<<<dim3(D_MODEL / 64, MBLK), 128, 0, stream>>>(
      attnb, wo, bo, out);
}
